// RecursiveEncoder_24068996726964
// MI455X (gfx1250) — matrix operand order repaired and checked by reading, not compiled, not run on hardware
//
#include <hip/hip_runtime.h>
#include <stdint.h>

// ---------------------------------------------------------------------------
// Problem constants (match reference)
// ---------------------------------------------------------------------------
#define D_   128
#define M_   8192
#define L_   5
#define K_   8
#define KL_  4
#define G_   4096

// Rows (tree nodes / statements) per workgroup: two 16-row WMMA M-tiles.
// Each B fragment fetched from L2 is reused by both M-tiles -> halves the
// dominant L2->VGPR weight traffic vs. one M-tile per workgroup.
#define ROWS_  32

typedef __attribute__((ext_vector_type(16))) __bf16        v16bf;
typedef __attribute__((ext_vector_type(8)))  float         v8f;
typedef __attribute__((ext_vector_type(4)))  unsigned int  u32x4;

union Frag16 {
    struct { u32x4 lo, hi; } u;
    v16bf v;
};

// ---------------------------------------------------------------------------
// WMMA fragment helpers (bf16 A/B, f32 C/D), layouts per CDNA5 ISA 7.12.2
// ---------------------------------------------------------------------------

// A: 16x32 (MxK) tile starting at K = kk*32, row-major source with stride ld.
// lanes 0-15: M=lane,    K = kk*32 + {0..7, 16..23}
// lanes16-31: M=lane-16, K = kk*32 + {8..15, 24..31}
__device__ inline v16bf load_a_frag(const __bf16* xs, int ld, int kk, int lane) {
    int m  = lane & 15;
    int kh = (lane >> 4) << 3;          // 0 or 8
    const __bf16* p = xs + m * ld + kk * 32 + kh;
    Frag16 f;
    f.u.lo = *(const u32x4*)(p);
    f.u.hi = *(const u32x4*)(p + 16);
    return f.v;
}

// B: 32x16 (KxN) tile: lane l holds column N = nn*16 + l%16, K = kk*32 + {0..7,16..23} (+8 for lanes 16-31).
__device__ inline v16bf load_b_frag(const __bf16* W, int ldn, int kk, int nn, int lane) {
    const __bf16* p = W + (size_t)(kk * 32 + ((lane >> 4) << 3)) * ldn + nn * 16 + (lane & 15);
    v16bf v;
    #pragma unroll
    for (int s = 0; s < 16; ++s) v[s] = p[(size_t)(s < 8 ? s : s + 8) * ldn];
    return v;
}

__device__ inline v8f wmma_bf16(v16bf a, v16bf b, v8f c) {
    return __builtin_amdgcn_wmma_f32_16x16x32_bf16(
        /*neg_a=*/false, a, /*neg_b=*/false, b,
        /*c_mod=*/(short)0, c, /*reuse_a=*/false, /*reuse_b=*/false);
}

// D: lane l holds N = l%16, rows M = r + 8*(l/16), r = 0..7 (v8f).
__device__ inline void store_d_lds(__bf16* out, int ld, int nn, v8f acc,
                                   const float* bias, bool relu, int lane) {
    int n  = lane & 15;
    int mb = (lane >> 4) << 3;
    float bv = bias[nn * 16 + n];
    #pragma unroll
    for (int r = 0; r < 8; ++r) {
        float v = acc[r] + bv;
        if (relu) v = v > 0.f ? v : 0.f;
        out[(mb + r) * ld + nn * 16 + n] = (__bf16)v;
    }
}

// ---------------------------------------------------------------------------
// Utility kernels
// ---------------------------------------------------------------------------
__global__ void f32_to_bf16_kernel(const float* __restrict__ in,
                                   __bf16* __restrict__ out, int n) {
    int i = blockIdx.x * blockDim.x + threadIdx.x;
    if (i < n) out[i] = (__bf16)in[i];
}

__global__ void gather_leaves_kernel(const int* __restrict__ leaf_ids,
                                     const float* __restrict__ emb,
                                     __bf16* __restrict__ h0) {
    int i = blockIdx.x * blockDim.x + threadIdx.x;   // over M*D
    if (i < M_ * D_) {
        int m = i >> 7, c = i & 127;
        h0[i] = (__bf16)emb[(size_t)leaf_ids[m] * D_ + c];
    }
}

// Counting-sort of items by fid into ROWS_-aligned groups; pads with -1.
// One block, 256 threads. Output order is nondeterministic within a group but
// consumers write per-item results, so final output is deterministic.
__global__ void group_kernel(const int* __restrict__ fid, int n, int F,
                             int* __restrict__ order, int* __restrict__ tile_fid,
                             int maxTiles) {
    __shared__ int cnt[8], off[8], cur[8];
    int tid = threadIdx.x;
    if (tid < F) { cnt[tid] = 0; cur[tid] = 0; }
    __syncthreads();
    for (int i = tid; i < n; i += blockDim.x) atomicAdd(&cnt[fid[i]], 1);
    __syncthreads();
    if (tid == 0) {
        int run = 0;
        for (int f = 0; f < F; ++f) {
            off[f] = run;
            run += ((cnt[f] + ROWS_ - 1) / ROWS_) * ROWS_;
        }
    }
    __syncthreads();
    int padded = maxTiles * ROWS_;
    for (int i = tid; i < padded; i += blockDim.x) order[i] = -1;
    if (tid == 0) {
        for (int t = 0; t < maxTiles; ++t) tile_fid[t] = 0;
        for (int f = 0; f < F; ++f) {
            int t0 = off[f] / ROWS_;
            int t1 = (off[f] + ((cnt[f] + ROWS_ - 1) / ROWS_) * ROWS_) / ROWS_;
            for (int t = t0; t < t1; ++t) tile_fid[t] = f;
        }
    }
    __syncthreads();
    for (int i = tid; i < n; i += blockDim.x) {
        int f = fid[i];
        int p = off[f] + atomicAdd(&cur[f], 1);
        order[p] = i;
    }
}

// ---------------------------------------------------------------------------
// Tree level: per 32-node tile, x=[h[l];h[r]] (32x256) -> relu(x W1 + b1)
//   (32x512) -> relu(. W2 + b2) (32x256) -> . W3 + b3 (32x128) -> h_next
// 128 threads = 4 waves per block; each wave runs 2 M-tiles per B fragment.
// ---------------------------------------------------------------------------
#define XS_LD  264     // 256 + 8 (bank-conflict pad)
#define A1_LD  520     // 512 + 8
#define A2_LD  264

__global__ __launch_bounds__(128)
void level_kernel(const __bf16* __restrict__ h, __bf16* __restrict__ hn,
                  const int* __restrict__ order, const int* __restrict__ tile_fid,
                  const int* __restrict__ left, const int* __restrict__ right,
                  const __bf16* __restrict__ W1, const float* __restrict__ b1,
                  const __bf16* __restrict__ W2, const float* __restrict__ b2,
                  const __bf16* __restrict__ W3, const float* __restrict__ b3) {
    __shared__ __bf16 xs[ROWS_ * XS_LD];
    __shared__ __bf16 a1s[ROWS_ * A1_LD];
    __shared__ __bf16 a2s[ROWS_ * A2_LD];
    __shared__ int nodes[ROWS_];

    int t    = blockIdx.x;
    int tid  = threadIdx.x;
    int lane = tid & 31;
    int wid  = tid >> 5;
    int fid  = tile_fid[t];

    if (tid < ROWS_) nodes[tid] = order[t * ROWS_ + tid];
    __syncthreads();

    // Gather x = concat(h[left], h[right]) : ROWS_ rows x 32 chunks of 8 bf16
    for (int ch = tid; ch < ROWS_ * 32; ch += 128) {
        int r = ch >> 5, c = ch & 31;
        int node = nodes[r];
        u32x4 val = 0;
        if (node >= 0) {
            int src = (c < 16) ? left[node] : right[node];
            val = *(const u32x4*)(h + (size_t)src * D_ + (c & 15) * 8);
        }
        *(u32x4*)(xs + r * XS_LD + c * 8) = val;
    }
    __syncthreads();

    // Layer 1: [32x256] x [256x512] + b1, relu
    {
        const __bf16* w = W1 + (size_t)fid * 256 * 512;
        const float*  b = b1 + fid * 512;
        for (int j = 0; j < 8; ++j) {
            int nn = wid * 8 + j;
            v8f acc0 = {}, acc1 = {};
            #pragma unroll
            for (int kk = 0; kk < 8; ++kk) {
                v16bf bm = load_b_frag(w, 512, kk, nn, lane);
                v16bf a0 = load_a_frag(xs, XS_LD, kk, lane);
                v16bf a1 = load_a_frag(xs + 16 * XS_LD, XS_LD, kk, lane);
                acc0 = wmma_bf16(a0, bm, acc0);
                acc1 = wmma_bf16(a1, bm, acc1);
            }
            store_d_lds(a1s,               A1_LD, nn, acc0, b, true, lane);
            store_d_lds(a1s + 16 * A1_LD,  A1_LD, nn, acc1, b, true, lane);
        }
    }
    __syncthreads();

    // Layer 2: [32x512] x [512x256] + b2, relu
    {
        const __bf16* w = W2 + (size_t)fid * 512 * 256;
        const float*  b = b2 + fid * 256;
        for (int j = 0; j < 4; ++j) {
            int nn = wid * 4 + j;
            v8f acc0 = {}, acc1 = {};
            #pragma unroll
            for (int kk = 0; kk < 16; ++kk) {
                v16bf bm = load_b_frag(w, 256, kk, nn, lane);
                v16bf a0 = load_a_frag(a1s, A1_LD, kk, lane);
                v16bf a1 = load_a_frag(a1s + 16 * A1_LD, A1_LD, kk, lane);
                acc0 = wmma_bf16(a0, bm, acc0);
                acc1 = wmma_bf16(a1, bm, acc1);
            }
            store_d_lds(a2s,              A2_LD, nn, acc0, b, true, lane);
            store_d_lds(a2s + 16 * A2_LD, A2_LD, nn, acc1, b, true, lane);
        }
    }
    __syncthreads();

    // Layer 3: [32x256] x [256x128] + b3 -> h_next (bf16)
    {
        const __bf16* w = W3 + (size_t)fid * 256 * 128;
        const float*  b = b3 + fid * 128;
        for (int j = 0; j < 2; ++j) {
            int nn = wid * 2 + j;
            v8f acc0 = {}, acc1 = {};
            #pragma unroll
            for (int kk = 0; kk < 8; ++kk) {
                v16bf bm = load_b_frag(w, 128, kk, nn, lane);
                v16bf a0 = load_a_frag(a2s, A2_LD, kk, lane);
                v16bf a1 = load_a_frag(a2s + 16 * A2_LD, A2_LD, kk, lane);
                acc0 = wmma_bf16(a0, bm, acc0);
                acc1 = wmma_bf16(a1, bm, acc1);
            }
            int n  = lane & 15;
            int mb = (lane >> 4) << 3;
            float bv = b[nn * 16 + n];
            #pragma unroll
            for (int r = 0; r < 8; ++r) {
                int n0 = nodes[mb + r];
                if (n0 >= 0)
                    hn[(size_t)n0 * D_ + nn * 16 + n] = (__bf16)(acc0[r] + bv);
                int n1 = nodes[16 + mb + r];
                if (n1 >= 0)
                    hn[(size_t)n1 * D_ + nn * 16 + n] = (__bf16)(acc1[r] + bv);
            }
        }
    }
}

// ---------------------------------------------------------------------------
// Logic head: x=[hL[gl];hL[gr]] (32x256) -> relu(x Wl1 + bl1) (32x512)
//   -> . Wl2 + bl2 (32x128) -> out (f32)
// ---------------------------------------------------------------------------
__global__ __launch_bounds__(128)
void logic_kernel(const __bf16* __restrict__ h, float* __restrict__ out,
                  const int* __restrict__ order, const int* __restrict__ tile_fid,
                  const int* __restrict__ gl, const int* __restrict__ gr,
                  const __bf16* __restrict__ Wl1, const float* __restrict__ bl1,
                  const __bf16* __restrict__ Wl2, const float* __restrict__ bl2) {
    __shared__ __bf16 xs[ROWS_ * XS_LD];
    __shared__ __bf16 g1s[ROWS_ * A1_LD];
    __shared__ int nodes[ROWS_];

    int t    = blockIdx.x;
    int tid  = threadIdx.x;
    int lane = tid & 31;
    int wid  = tid >> 5;
    int fid  = tile_fid[t];

    if (tid < ROWS_) nodes[tid] = order[t * ROWS_ + tid];
    __syncthreads();

    for (int ch = tid; ch < ROWS_ * 32; ch += 128) {
        int r = ch >> 5, c = ch & 31;
        int node = nodes[r];
        u32x4 val = 0;
        if (node >= 0) {
            int src = (c < 16) ? gl[node] : gr[node];
            val = *(const u32x4*)(h + (size_t)src * D_ + (c & 15) * 8);
        }
        *(u32x4*)(xs + r * XS_LD + c * 8) = val;
    }
    __syncthreads();

    // Layer 1: [32x256] x [256x512] + bl1, relu
    {
        const __bf16* w = Wl1 + (size_t)fid * 256 * 512;
        const float*  b = bl1 + fid * 512;
        for (int j = 0; j < 8; ++j) {
            int nn = wid * 8 + j;
            v8f acc0 = {}, acc1 = {};
            #pragma unroll
            for (int kk = 0; kk < 8; ++kk) {
                v16bf bm = load_b_frag(w, 512, kk, nn, lane);
                v16bf a0 = load_a_frag(xs, XS_LD, kk, lane);
                v16bf a1 = load_a_frag(xs + 16 * XS_LD, XS_LD, kk, lane);
                acc0 = wmma_bf16(a0, bm, acc0);
                acc1 = wmma_bf16(a1, bm, acc1);
            }
            store_d_lds(g1s,              A1_LD, nn, acc0, b, true, lane);
            store_d_lds(g1s + 16 * A1_LD, A1_LD, nn, acc1, b, true, lane);
        }
    }
    __syncthreads();

    // Layer 2: [32x512] x [512x128] + bl2 -> out (f32)
    {
        const __bf16* w = Wl2 + (size_t)fid * 512 * 128;
        const float*  b = bl2 + fid * 128;
        for (int j = 0; j < 2; ++j) {
            int nn = wid * 2 + j;
            v8f acc0 = {}, acc1 = {};
            #pragma unroll
            for (int kk = 0; kk < 16; ++kk) {
                v16bf bm = load_b_frag(w, 128, kk, nn, lane);
                v16bf a0 = load_a_frag(g1s, A1_LD, kk, lane);
                v16bf a1 = load_a_frag(g1s + 16 * A1_LD, A1_LD, kk, lane);
                acc0 = wmma_bf16(a0, bm, acc0);
                acc1 = wmma_bf16(a1, bm, acc1);
            }
            int n  = lane & 15;
            int mb = (lane >> 4) << 3;
            float bv = b[nn * 16 + n];
            #pragma unroll
            for (int r = 0; r < 8; ++r) {
                int n0 = nodes[mb + r];
                if (n0 >= 0)
                    out[(size_t)n0 * D_ + nn * 16 + n] = acc0[r] + bv;
                int n1 = nodes[16 + mb + r];
                if (n1 >= 0)
                    out[(size_t)n1 * D_ + nn * 16 + n] = acc1[r] + bv;
            }
        }
    }
}

// ---------------------------------------------------------------------------
// Host launch
// ---------------------------------------------------------------------------
extern "C" void kernel_launch(void* const* d_in, const int* in_sizes, int n_in,
                              void* d_out, int out_size, void* d_ws, size_t ws_size,
                              hipStream_t stream) {
    (void)in_sizes; (void)n_in; (void)out_size; (void)ws_size;

    const int*   leaf_ids  = (const int*)d_in[0];
    const int*   left_idx  = (const int*)d_in[1];
    const int*   right_idx = (const int*)d_in[2];
    const int*   nf_fid    = (const int*)d_in[3];
    const int*   gt_left   = (const int*)d_in[4];
    const int*   gt_right  = (const int*)d_in[5];
    const int*   lf_fid    = (const int*)d_in[6];
    const float* emb       = (const float*)d_in[7];
    const float* W1        = (const float*)d_in[8];
    const float* b1        = (const float*)d_in[9];
    const float* W2        = (const float*)d_in[10];
    const float* b2        = (const float*)d_in[11];
    const float* W3        = (const float*)d_in[12];
    const float* b3        = (const float*)d_in[13];
    const float* Wl1       = (const float*)d_in[14];
    const float* bl1       = (const float*)d_in[15];
    const float* Wl2       = (const float*)d_in[16];
    const float* bl2       = (const float*)d_in[17];
    float* out = (float*)d_out;

    // Workspace carve (256B aligned chunks)
    uintptr_t ws = (uintptr_t)d_ws;
    auto alloc = [&](size_t bytes) -> void* {
        void* p = (void*)ws;
        ws += (bytes + 255) & ~(size_t)255;
        return p;
    };
    const int nW1  = K_ * 256 * 512;
    const int nW2  = K_ * 512 * 256;
    const int nW3  = K_ * 256 * 128;
    const int nWl1 = KL_ * 256 * 512;
    const int nWl2 = KL_ * 512 * 128;

    __bf16* wW1  = (__bf16*)alloc((size_t)nW1 * 2);
    __bf16* wW2  = (__bf16*)alloc((size_t)nW2 * 2);
    __bf16* wW3  = (__bf16*)alloc((size_t)nW3 * 2);
    __bf16* wWl1 = (__bf16*)alloc((size_t)nWl1 * 2);
    __bf16* wWl2 = (__bf16*)alloc((size_t)nWl2 * 2);
    __bf16* hA   = (__bf16*)alloc((size_t)M_ * D_ * 2);
    __bf16* hB   = (__bf16*)alloc((size_t)M_ * D_ * 2);
    const int levelTiles = M_ / ROWS_ + K_;      // 264 (ROWS_-aligned fid groups)
    const int logicTiles = G_ / ROWS_ + KL_;     // 132
    int* order    = (int*)alloc((size_t)levelTiles * ROWS_ * 4);
    int* tile_fid = (int*)alloc((size_t)levelTiles * 4);

    // 1) Weights -> bf16 (L2-resident working set, ~5 MB)
    f32_to_bf16_kernel<<<(nW1  + 255) / 256, 256, 0, stream>>>(W1,  wW1,  nW1);
    f32_to_bf16_kernel<<<(nW2  + 255) / 256, 256, 0, stream>>>(W2,  wW2,  nW2);
    f32_to_bf16_kernel<<<(nW3  + 255) / 256, 256, 0, stream>>>(W3,  wW3,  nW3);
    f32_to_bf16_kernel<<<(nWl1 + 255) / 256, 256, 0, stream>>>(Wl1, wWl1, nWl1);
    f32_to_bf16_kernel<<<(nWl2 + 255) / 256, 256, 0, stream>>>(Wl2, wWl2, nWl2);

    // 2) Leaf embeddings -> bf16 h0
    gather_leaves_kernel<<<(M_ * D_ + 255) / 256, 256, 0, stream>>>(leaf_ids, emb, hA);

    // 3) Five levels, fid-grouped WMMA MLPs, ping-pong h buffers
    __bf16* cur = hA;
    __bf16* nxt = hB;
    for (int lvl = 0; lvl < L_; ++lvl) {
        group_kernel<<<1, 256, 0, stream>>>(nf_fid + lvl * M_, M_, K_,
                                            order, tile_fid, levelTiles);
        level_kernel<<<levelTiles, 128, 0, stream>>>(
            cur, nxt, order, tile_fid,
            left_idx + lvl * M_, right_idx + lvl * M_,
            wW1, b1, wW2, b2, wW3, b3);
        __bf16* t = cur; cur = nxt; nxt = t;
    }

    // 4) Logic head -> f32 output
    group_kernel<<<1, 256, 0, stream>>>(lf_fid, G_, KL_, order, tile_fid, logicTiles);
    logic_kernel<<<logicTiles, 128, 0, stream>>>(
        cur, out, order, tile_fid, gt_left, gt_right, wWl1, bl1, wWl2, bl2);
}
